// ARTAttrEncoderGCN_32727650796179
// MI455X (gfx1250) — matrix operand order repaired and checked by reading, not compiled, not run on hardware
//
#include <hip/hip_runtime.h>
#include <hip/hip_bf16.h>

#define HID 128
#define GEMM_BLOCK 256
#define GEMM_ROWS 64           // rows of output per block
#define LDH 136                // padded f16 LDS row stride (272B -> distinct banks)

typedef __attribute__((ext_vector_type(16))) _Float16 v16h;
typedef __attribute__((ext_vector_type(8)))  float    v8f;

// ---------------------------------------------------------------------------
// Generic zero fill
// ---------------------------------------------------------------------------
__global__ void zero_kernel(float* __restrict__ p, int n) {
    int i = blockIdx.x * blockDim.x + threadIdx.x;
    if (i < n) p[i] = 0.0f;
}

// ---------------------------------------------------------------------------
// ReLU in place
// ---------------------------------------------------------------------------
__global__ void relu_kernel(float* __restrict__ p, int n) {
    int i = blockIdx.x * blockDim.x + threadIdx.x;
    if (i < n) p[i] = fmaxf(p[i], 0.0f);
}

// ---------------------------------------------------------------------------
// Edge embed + scatter: agg[dst[e]] += edge_attr[e] @ W_edge
// One wave (32 lanes) per edge; lane handles 4 of 128 output columns.
// ---------------------------------------------------------------------------
__global__ __launch_bounds__(256)
void edge_embed_scatter(const float* __restrict__ edge_attr,  // [E x 8]
                        const int*   __restrict__ dst,        // [E]
                        const float* __restrict__ W_edge,     // [8 x 128]
                        float*       __restrict__ agg,        // [N x 128]
                        int E) {
    __shared__ float w[8 * HID];
    for (int i = threadIdx.x; i < 8 * HID; i += blockDim.x) w[i] = W_edge[i];
    __syncthreads();

    int lane = threadIdx.x & 31;
    int e = blockIdx.x * 8 + (threadIdx.x >> 5);
    if (e >= E) return;

    float ea[8];
#pragma unroll
    for (int k = 0; k < 8; ++k) ea[k] = edge_attr[e * 8 + k];
    int d = dst[e];

#pragma unroll
    for (int c4 = 0; c4 < 4; ++c4) {
        int col = lane + c4 * 32;
        float s = 0.0f;
#pragma unroll
        for (int k = 0; k < 8; ++k) s += ea[k] * w[k * HID + col];
        atomicAdd(&agg[(long)d * HID + col], s);
    }
}

// ---------------------------------------------------------------------------
// Node embed: h = relu(node_attr @ W_node + agg + b_embed)
// Thread per (node, col).
// ---------------------------------------------------------------------------
__global__ __launch_bounds__(256)
void node_embed(const float* __restrict__ node_attr,  // [N x 16]
                const float* __restrict__ agg,        // [N x 128]
                const float* __restrict__ W_node,     // [16 x 128]
                const float* __restrict__ b_embed,    // [128]
                float*       __restrict__ h,          // [N x 128]
                int N) {
    __shared__ float w[16 * HID];
    for (int i = threadIdx.x; i < 16 * HID; i += blockDim.x) w[i] = W_node[i];
    __syncthreads();

    int idx = blockIdx.x * blockDim.x + threadIdx.x;
    if (idx >= N * HID) return;
    int n = idx >> 7, c = idx & 127;
    float s = agg[idx] + b_embed[c];
#pragma unroll
    for (int k = 0; k < 16; ++k) s += node_attr[n * 16 + k] * w[k * HID + c];
    h[idx] = fmaxf(s, 0.0f);
}

// ---------------------------------------------------------------------------
// WMMA GEMM: Y[rows x 128] = X[rows x 128] @ W[128 x 128] + bias (+ relu)
// Block = 256 threads (8 wave32). Block tile: 64 rows x 128 cols.
// Wave w: row tile (w & 3), col tiles 4*(w>>2) .. +3.
// A and W^T staged in LDS as f16 (stride LDH=136 halves -> no bank conflicts
// on fragment reads). K=128 done as 4x v_wmma_f32_16x16x32_f16, f32 accum.
// Fragment packing per CDNA5 ISA 7.12.2:
//   A (16x32 f16): vgpr j -> lo lanes K = (j>>2)*16 + 2*(j&3); hi lanes +8
//   B (32x16 f16): vgpr j -> lo lanes K = (j>>2)*16 + 2*(j&3);                 hi lanes +8
// RELU is a template parameter; full tiles take an unconditional store path.
// ---------------------------------------------------------------------------
template <int DO_RELU>
__global__ __launch_bounds__(GEMM_BLOCK)
void gemm_bias_wmma(const float* __restrict__ X,     // [rows x 128]
                    const float* __restrict__ W,     // [128 x 128]
                    const float* __restrict__ bias,  // [128]
                    float*       __restrict__ Y,     // [rows x 128]
                    int rows) {
    __shared__ _Float16 ldsA[GEMM_ROWS * LDH];  // A tile, f16
    __shared__ _Float16 ldsW[HID * LDH];        // W transposed: ldsW[n][k]

    const int tid = threadIdx.x;
    const int row0 = blockIdx.x * GEMM_ROWS;

    // Stage W transposed (coalesced global reads over n)
    for (int i = tid; i < HID * HID; i += GEMM_BLOCK) {
        int k = i >> 7, n = i & 127;
        ldsW[n * LDH + k] = (_Float16)W[i];
    }
    // Stage A tile (guarded)
    for (int i = tid; i < GEMM_ROWS * HID; i += GEMM_BLOCK) {
        int r = i >> 7, c = i & 127;
        int gr = row0 + r;
        float v = (gr < rows) ? X[(long)gr * HID + c] : 0.0f;
        ldsA[r * LDH + c] = (_Float16)v;
    }
    __syncthreads();

    const int wave = tid >> 5;
    const int lane = tid & 31;
    const int hi   = lane >> 4;   // half-wave select
    const int ln   = lane & 15;
    const int m0   = (wave & 3) * 16;       // wave's row tile within block
    const int ntb  = (wave >> 2) * 4;       // first col tile for this wave
    const bool full = (row0 + GEMM_ROWS) <= rows;   // uniform across block

    // Preload the 4 A fragments (K = 0..127) once; reuse over 4 col tiles.
    v16h afrag[4];
#pragma unroll
    for (int kk = 0; kk < 4; ++kk) {
#pragma unroll
        for (int j = 0; j < 8; ++j) {
            int kb = kk * 32 + ((j >> 2) * 16) + hi * 8 + 2 * (j & 3);
            afrag[kk][2 * j]     = ldsA[(m0 + ln) * LDH + kb];
            afrag[kk][2 * j + 1] = ldsA[(m0 + ln) * LDH + kb + 1];
        }
    }

#pragma unroll
    for (int nt = 0; nt < 4; ++nt) {
        const int n0 = (ntb + nt) * 16;
        v8f c = {};
#pragma unroll
        for (int kk = 0; kk < 4; ++kk) {
            v16h b;
#pragma unroll
            for (int j = 0; j < 8; ++j) {
                int kb = kk * 32 + ((j >> 2) * 16) + hi * 8 + 2 * (j & 3);
                b[2 * j]     = ldsW[(n0 + ln) * LDH + kb];
                b[2 * j + 1] = ldsW[(n0 + ln) * LDH + kb + 1];
            }
            c = __builtin_amdgcn_wmma_f32_16x16x32_f16(
                    /*neg_a=*/false, afrag[kk], /*neg_b=*/false, b,
                    /*c_mod=*/(short)0, c, /*reuse_a=*/false, /*reuse_b=*/false);
        }
        // C/D layout: vgpr r -> row m0 + r + 8*hi, col n0 + ln
        const float bv = bias[n0 + ln];
        float* yp = Y + (long)(row0 + m0 + 8 * hi) * HID + n0 + ln;
        if (full) {
            // Fast path: unconditional strided stores, no per-element guards.
#pragma unroll
            for (int r = 0; r < 8; ++r) {
                float v = c[r] + bv;
                if (DO_RELU) v = fmaxf(v, 0.0f);
                yp[(long)r * HID] = v;
            }
        } else {
#pragma unroll
            for (int r = 0; r < 8; ++r) {
                int gm = row0 + m0 + r + 8 * hi;
                if (gm < rows) {
                    float v = c[r] + bv;
                    if (DO_RELU) v = fmaxf(v, 0.0f);
                    yp[(long)r * HID] = v;
                }
            }
        }
    }
}

// ---------------------------------------------------------------------------
// SpMM scatter: out[src[e]] += adj_value[e] * hw[dst[e]]
// One wave per edge; lane owns a float4 (coalesced 512B row gather).
// ---------------------------------------------------------------------------
__global__ __launch_bounds__(256)
void spmm_scatter(const float* __restrict__ hw,    // [N x 128]
                  const int*   __restrict__ src,   // [E]
                  const int*   __restrict__ dst,   // [E]
                  const float* __restrict__ aval,  // [E]
                  float*       __restrict__ out,   // [N x 128]
                  int E) {
    int lane = threadIdx.x & 31;
    int e = blockIdx.x * 8 + (threadIdx.x >> 5);
    if (e >= E) return;
    int s = src[e], d = dst[e];
    float v = aval[e];
    const float4* row = (const float4*)(hw + (long)d * HID);
    float4 m = row[lane];
    float* o = out + (long)s * HID + lane * 4;
    atomicAdd(o + 0, v * m.x);
    atomicAdd(o + 1, v * m.y);
    atomicAdd(o + 2, v * m.z);
    atomicAdd(o + 3, v * m.w);
}

// ---------------------------------------------------------------------------
// Global add pool: fp[batch[n]] += h[n]
// ---------------------------------------------------------------------------
__global__ __launch_bounds__(256)
void pool_kernel(const float* __restrict__ h, const int* __restrict__ batch,
                 float* __restrict__ fp, int N) {
    int idx = blockIdx.x * blockDim.x + threadIdx.x;
    if (idx >= N * HID) return;
    int n = idx >> 7, c = idx & 127;
    atomicAdd(&fp[(long)batch[n] * HID + c], h[idx]);
}

// ---------------------------------------------------------------------------
// Final predictor row: out[g] = z[g] . Wp2 + bp2
// Block (128 threads) per graph.
// ---------------------------------------------------------------------------
__global__ __launch_bounds__(128)
void pred_final(const float* __restrict__ z,    // [G x 128] (already relu'd)
                const float* __restrict__ Wp2,  // [128 x 1]
                const float* __restrict__ bp2,  // [1]
                float* __restrict__ out, int G) {
    __shared__ float red[128];
    int g = blockIdx.x;
    int t = threadIdx.x;
    red[t] = z[(long)g * HID + t] * Wp2[t];
    __syncthreads();
    for (int s = 64; s > 0; s >>= 1) {
        if (t < s) red[t] += red[t + s];
        __syncthreads();
    }
    if (t == 0) out[g] = red[0] + bp2[0];
}

// ---------------------------------------------------------------------------
// Host-side launch sequence
// ---------------------------------------------------------------------------
extern "C" void kernel_launch(void* const* d_in, const int* in_sizes, int n_in,
                              void* d_out, int out_size, void* d_ws, size_t ws_size,
                              hipStream_t stream) {
    const float* node_attr  = (const float*)d_in[0];   // [N x 16]
    const float* edge_attr  = (const float*)d_in[1];   // [E x 8]
    const int*   edge_index = (const int*)  d_in[2];   // [2 x E]
    const int*   adj_index  = (const int*)  d_in[3];   // [2 x E]
    const float* adj_value  = (const float*)d_in[4];   // [E]
    const int*   batch      = (const int*)  d_in[5];   // [N]
    const float* W_node = (const float*)d_in[6];
    const float* W_edge = (const float*)d_in[7];
    const float* b_embed= (const float*)d_in[8];
    const float* W1 = (const float*)d_in[9];   const float* b1 = (const float*)d_in[10];
    const float* W2 = (const float*)d_in[11];  const float* b2 = (const float*)d_in[12];
    const float* W3 = (const float*)d_in[13];  const float* b3 = (const float*)d_in[14];
    const float* Wp1= (const float*)d_in[15];  const float* bp1= (const float*)d_in[16];
    const float* Wp2= (const float*)d_in[17];  const float* bp2= (const float*)d_in[18];
    float* out = (float*)d_out;

    const int N = in_sizes[0] / 16;
    const int E = in_sizes[1] / 8;
    const int G = out_size;         // [G x 1]
    const long NH = (long)N * HID;
    const long GH = (long)G * HID;

    // edge_index row 1 = dst for embedding aggregation
    const int* emb_dst = edge_index + E;
    const int* a_src = adj_index;       // row 0
    const int* a_dst = adj_index + E;   // row 1

    // Workspace carving (floats): 3 node buffers + fp + z
    float* ws   = (float*)d_ws;
    float* bufA = ws;                 // [N x 128]
    float* bufB = ws + NH;            // [N x 128]
    float* bufC = ws + 2 * NH;        // [N x 128]
    float* fp   = ws + 3 * NH;        // [G x 128]
    float* zbuf = fp + GH;            // [G x 128]

    const int NHb = (int)NH;
    const int zgrid  = (NHb + 255) / 256;
    const int egrid  = (E + 7) / 8;
    const int ggrid  = (N + GEMM_ROWS - 1) / GEMM_ROWS;
    const int pgrid  = (G + GEMM_ROWS - 1) / GEMM_ROWS;

    // ---- attr embedder ----
    zero_kernel<<<zgrid, 256, 0, stream>>>(bufB, NHb);                      // agg
    edge_embed_scatter<<<egrid, 256, 0, stream>>>(edge_attr, emb_dst, W_edge, bufB, E);
    node_embed<<<zgrid, 256, 0, stream>>>(node_attr, bufB, W_node, b_embed, bufA, N);

    // ---- GCN layer 1: h=bufA -> bufC ----
    gemm_bias_wmma<0><<<ggrid, GEMM_BLOCK, 0, stream>>>(bufA, W1, b1, bufB, N);
    zero_kernel<<<zgrid, 256, 0, stream>>>(bufC, NHb);
    spmm_scatter<<<egrid, 256, 0, stream>>>(bufB, a_src, a_dst, adj_value, bufC, E);
    relu_kernel<<<zgrid, 256, 0, stream>>>(bufC, NHb);

    // ---- GCN layer 2: h=bufC -> bufA ----
    gemm_bias_wmma<0><<<ggrid, GEMM_BLOCK, 0, stream>>>(bufC, W2, b2, bufB, N);
    zero_kernel<<<zgrid, 256, 0, stream>>>(bufA, NHb);
    spmm_scatter<<<egrid, 256, 0, stream>>>(bufB, a_src, a_dst, adj_value, bufA, E);
    relu_kernel<<<zgrid, 256, 0, stream>>>(bufA, NHb);

    // ---- GCN layer 3: h=bufA -> bufC ----
    gemm_bias_wmma<0><<<ggrid, GEMM_BLOCK, 0, stream>>>(bufA, W3, b3, bufB, N);
    zero_kernel<<<zgrid, 256, 0, stream>>>(bufC, NHb);
    spmm_scatter<<<egrid, 256, 0, stream>>>(bufB, a_src, a_dst, adj_value, bufC, E);
    relu_kernel<<<zgrid, 256, 0, stream>>>(bufC, NHb);

    // ---- global add pool ----
    zero_kernel<<<((int)GH + 255) / 256, 256, 0, stream>>>(fp, (int)GH);
    pool_kernel<<<zgrid, 256, 0, stream>>>(bufC, batch, fp, N);

    // ---- predictor MLP ----
    gemm_bias_wmma<1><<<pgrid, GEMM_BLOCK, 0, stream>>>(fp, Wp1, bp1, zbuf, G);
    pred_final<<<G, 128, 0, stream>>>(zbuf, Wp2, bp2, out, G);
}
